// GraphormerNodeEncoder_44865228374491
// MI455X (gfx1250) — matrix operand order repaired and checked by reading, not compiled, not run on hardware
//
#include <hip/hip_runtime.h>

// ---------------------------------------------------------------------------
// Graphormer node encoder for MI455X (gfx1250).
// GEMMs: v_wmma_f32_16x16x32_f16, 2x2 fragment wave tiles (32x32/wave),
//        double-buffered LDS, global_load_b128 staged loads, software
//        pipelined, v_cvt_pk_rtz_f16_f32 packing for f32->f16 staging.
// Edge softmax: wave-per-edge, wave32 shfl reductions, f32 global atomics.
// ---------------------------------------------------------------------------

#define Nn   20000
#define Ee   320000
#define ETOT (Ee + Nn)     // self loops appended
#define INC  128
#define Dd   256
#define Hh   8
#define DHh  32
#define Ll   2
#define Bb   64
#define DFFc 512

typedef __attribute__((ext_vector_type(16))) _Float16 v16h;
typedef __attribute__((ext_vector_type(8)))  float    v8f;
typedef __attribute__((ext_vector_type(2)))  __fp16   fp16x2;

// single-instruction pack: v_cvt_pk_rtz_f16_f32
static __device__ __forceinline__ unsigned pack2(float a, float b) {
  union { fp16x2 h; unsigned u; } x;
  x.h = __builtin_amdgcn_cvt_pkrtz(a, b);
  return x.u;
}

// ------------------------------- utility kernels ---------------------------

__global__ __launch_bounds__(256) void fill_f32(float* p, float v, int n) {
  int i = blockIdx.x * 256 + threadIdx.x;
  if (i < n) p[i] = v;
}

__global__ __launch_bounds__(256) void f32_to_f16(const float* __restrict__ s,
                                                  _Float16* __restrict__ d, int n) {
  int i = blockIdx.x * 256 + threadIdx.x;
  if (i < n) d[i] = (_Float16)s[i];
}

__global__ __launch_bounds__(256) void deg_count(const int* __restrict__ row,
                                                 float* __restrict__ deg, int e) {
  int i = blockIdx.x * 256 + threadIdx.x;
  if (i < e) atomicAdd(&deg[row[i]], 1.0f);
}

__global__ __launch_bounds__(256) void bucket_k(const float* __restrict__ deg,
                                                int* __restrict__ bucket, int n) {
  int i = blockIdx.x * 256 + threadIdx.x;
  if (i < n) {
    int b = (int)floorf(log2f(fmaxf(deg[i], 1.0f)));
    b = min(max(b, 1), Bb - 1);
    bucket[i] = b;
  }
}

// ------------------------------- WMMA GEMM ---------------------------------
// C[M x Nc] = A(f32, MxK row-major) @ W(f16, KxNc row-major) + bias, epilogue:
//   MODE 0: plain   MODE 1: + deg_emb[bucket[m]]   MODE 2: exact GeLU
// Block = 256 threads (8 waves), block tile 128x64, wave tile 32x32
// (2x2 fragments -> 4 WMMAs per K-step per wave), K step 32.
// Double-buffered LDS; next tile fetched to registers while WMMA runs.

template<int MODE>
__global__ __launch_bounds__(256) void gemm_wmma(
    const float* __restrict__ A, const _Float16* __restrict__ W,
    const float* __restrict__ bias, float* __restrict__ C,
    int M, int K, int Nc,
    const int* __restrict__ bucket, const float* __restrict__ dege)
{
  __shared__ _Float16 As[2][128 * 34];  // padded stride 34 halves
  __shared__ _Float16 Bs[2][64 * 34];   // transposed: Bs[n*34 + k]

  const int t     = threadIdx.x;
  const int lane  = t & 31;
  const int wave  = t >> 5;
  const int waveM = wave & 3;           // 4 M positions of 32 rows
  const int waveN = wave >> 2;          // 2 N positions of 32 cols
  const int bm = blockIdx.y * 128;
  const int bn = blockIdx.x * 64;

  // ---- staging geometry (branchless; A rows clamped to M-1) ----
  // Named scalar pointers (not arrays) so the backend infers global address
  // space -> global_load_b128 (FLAT loads would also tick DScnt and stall
  // the WMMA fragment waits).
  const int c4 = t & 7;                 // float4 slot within a 32-wide A row
  const int r0 = t >> 3;                // A row 0..31 (plus +32/+64/+96)
  int gm0 = bm + r0;       if (gm0 >= M) gm0 = M - 1;
  int gm1 = bm + r0 + 32;  if (gm1 >= M) gm1 = M - 1;
  int gm2 = bm + r0 + 64;  if (gm2 >= M) gm2 = M - 1;
  int gm3 = bm + r0 + 96;  if (gm3 >= M) gm3 = M - 1;
  const float4* ap0 = (const float4*)(A + (size_t)gm0 * K) + c4;
  const float4* ap1 = (const float4*)(A + (size_t)gm1 * K) + c4;
  const float4* ap2 = (const float4*)(A + (size_t)gm2 * K) + c4;
  const float4* ap3 = (const float4*)(A + (size_t)gm3 * K) + c4;
  const int rB = t >> 3;                // B k-row 0..31
  const int n0 = (t & 7) * 8;           // 8 consecutive n per thread
  const uint4* bp = (const uint4*)(W + (size_t)rB * Nc + bn + n0);
  const int apStep = 8;                 // 32 floats = 8 float4
  const int bpStep = (32 * Nc) >> 3;    // 32 rows of Nc halves, in uint4

  // ---- per-wave fragment K offsets (ISA 7.12.2 layouts) ----
  const int half = lane >> 4;
  const int mrow = waveM * 32 + (lane & 15);
  const int nc0  = waveN * 32 + (lane & 15);
  int aoffK[16], boffK[16];
  #pragma unroll
  for (int e = 0; e < 16; ++e) {
    int j = e >> 1, s = e & 1;
    aoffK[e] = ((j < 4) ? 2 * j : 2 * j + 8) + 8 * half + s;  // A: K-pair stripes
    boffK[e] = ((e < 8) ? e : e + 8) + 8 * half;              // B: same K stripes as A
  }

  v8f acc00 = {}, acc01 = {}, acc10 = {}, acc11 = {};
  const int nk = K >> 5;

  // preload first tile into registers
  float4 fa0 = *ap0, fa1 = *ap1, fa2 = *ap2, fa3 = *ap3;
  uint4 fb = *bp;

  int buf = 0;
  for (int s = 0; s < nk; ++s) {
    // ---- registers -> LDS[buf] (v_cvt_pk_rtz_f16_f32 packing) ----
    unsigned* aw0 = (unsigned*)&As[buf][(r0     ) * 34 + c4 * 4];
    aw0[0] = pack2(fa0.x, fa0.y);  aw0[1] = pack2(fa0.z, fa0.w);
    unsigned* aw1 = (unsigned*)&As[buf][(r0 + 32) * 34 + c4 * 4];
    aw1[0] = pack2(fa1.x, fa1.y);  aw1[1] = pack2(fa1.z, fa1.w);
    unsigned* aw2 = (unsigned*)&As[buf][(r0 + 64) * 34 + c4 * 4];
    aw2[0] = pack2(fa2.x, fa2.y);  aw2[1] = pack2(fa2.z, fa2.w);
    unsigned* aw3 = (unsigned*)&As[buf][(r0 + 96) * 34 + c4 * 4];
    aw3[0] = pack2(fa3.x, fa3.y);  aw3[1] = pack2(fa3.z, fa3.w);
    const _Float16* bh = (const _Float16*)&fb;
    #pragma unroll
    for (int i = 0; i < 8; ++i) Bs[buf][(n0 + i) * 34 + rB] = bh[i];
    __syncthreads();

    // ---- prefetch next tile to registers (overlaps WMMA below) ----
    if (s + 1 < nk) {
      ap0 += apStep; ap1 += apStep; ap2 += apStep; ap3 += apStep;
      bp  += bpStep;
      fa0 = *ap0; fa1 = *ap1; fa2 = *ap2; fa3 = *ap3;
      fb  = *bp;
    }

    // ---- fragments + 4 WMMAs ----
    v16h a0, a1, b0, b1;
    #pragma unroll
    for (int e = 0; e < 16; ++e) a0[e] = As[buf][mrow * 34 + aoffK[e]];
    #pragma unroll
    for (int e = 0; e < 16; ++e) a1[e] = As[buf][(mrow + 16) * 34 + aoffK[e]];
    #pragma unroll
    for (int e = 0; e < 16; ++e) b0[e] = Bs[buf][nc0 * 34 + boffK[e]];
    #pragma unroll
    for (int e = 0; e < 16; ++e) b1[e] = Bs[buf][(nc0 + 16) * 34 + boffK[e]];
    acc00 = __builtin_amdgcn_wmma_f32_16x16x32_f16(false, a0, false, b0, (short)0, acc00, false, false);
    acc01 = __builtin_amdgcn_wmma_f32_16x16x32_f16(false, a0, false, b1, (short)0, acc01, false, false);
    acc10 = __builtin_amdgcn_wmma_f32_16x16x32_f16(false, a1, false, b0, (short)0, acc10, false, false);
    acc11 = __builtin_amdgcn_wmma_f32_16x16x32_f16(false, a1, false, b1, (short)0, acc11, false, false);
    buf ^= 1;
  }

  // ---- epilogue: C/D layout — VGPR i: M = 8*(lane>=16)+i, N = lane&15 ----
  const int nl = lane & 15;
  #pragma unroll
  for (int mi = 0; mi < 2; ++mi) {
    #pragma unroll
    for (int ni = 0; ni < 2; ++ni) {
      v8f acc = (mi == 0) ? (ni == 0 ? acc00 : acc01)
                          : (ni == 0 ? acc10 : acc11);
      int gn = bn + waveN * 32 + ni * 16 + nl;
      #pragma unroll
      for (int i = 0; i < 8; ++i) {
        int gm = bm + waveM * 32 + mi * 16 + half * 8 + i;
        if (gm < M) {
          float v = acc[i] + bias[gn];
          if (MODE == 1) v += dege[(size_t)bucket[gm] * Nc + gn];
          if (MODE == 2) v = 0.5f * v * (1.0f + erff(v * 0.7071067811865475f));
          C[(size_t)gm * Nc + gn] = v;
        }
      }
    }
  }
}

// ------------------------------- edge kernels ------------------------------

__device__ __forceinline__ void atomic_max_f32(float* addr, float val) {
  if (val >= 0.0f) atomicMax((int*)addr, __float_as_int(val));
  else             atomicMin((unsigned int*)addr, __float_as_uint(val));
}

// pass 1: per-edge per-head scores + spatial bias, segment max (wave per edge)
__global__ __launch_bounds__(256) void edge_score(
    const int* __restrict__ row, const int* __restrict__ col,
    const float* __restrict__ Q, const float* __restrict__ Kx,
    const int* __restrict__ bucket, const float* __restrict__ sb,
    float* __restrict__ scores, float* __restrict__ mx, float scale)
{
  int e = blockIdx.x * 8 + (threadIdx.x >> 5);
  if (e >= ETOT) return;
  int lane = threadIdx.x & 31;
  int r, c;
  if (e < Ee) { r = row[e]; c = col[e]; } else { r = e - Ee; c = r; }
  const float* qb = Q  + (size_t)r * Dd;
  const float* kb = Kx + (size_t)c * Dd;
  float myscore = 0.0f;
  #pragma unroll
  for (int h = 0; h < Hh; ++h) {
    float p = qb[h * DHh + lane] * kb[h * DHh + lane];
    #pragma unroll
    for (int off = 16; off > 0; off >>= 1) p += __shfl_xor(p, off, 32);
    if (lane == h) myscore = p;
  }
  if (lane < Hh) {
    float s = myscore * scale + sb[bucket[r] * Hh + lane] + sb[bucket[c] * Hh + lane];
    scores[(size_t)e * Hh + lane] = s;
    atomic_max_f32(&mx[(size_t)r * Hh + lane], s);
  }
}

// pass 2: exp(score - max), segment sum (thread per (edge, head))
__global__ __launch_bounds__(256) void edge_exp(
    const int* __restrict__ row, const float* __restrict__ mx,
    float* __restrict__ scores, float* __restrict__ ss)
{
  int t = blockIdx.x * 256 + threadIdx.x;
  if (t >= ETOT * Hh) return;
  int e = t >> 3, hh = t & 7;
  int r = (e < Ee) ? row[e] : e - Ee;
  float es = expf(scores[t] - mx[(size_t)r * Hh + hh]);
  scores[t] = es;
  atomicAdd(&ss[(size_t)r * Hh + hh], es);
}

// pass 3: agg[r] += alpha * V[c] (wave per edge, lane per head-dim)
__global__ __launch_bounds__(256) void edge_agg(
    const int* __restrict__ row, const int* __restrict__ col,
    const float* __restrict__ V, const float* __restrict__ scores,
    const float* __restrict__ ss, float* __restrict__ agg)
{
  int e = blockIdx.x * 8 + (threadIdx.x >> 5);
  if (e >= ETOT) return;
  int lane = threadIdx.x & 31;
  int r, c;
  if (e < Ee) { r = row[e]; c = col[e]; } else { r = e - Ee; c = r; }
  #pragma unroll
  for (int h = 0; h < Hh; ++h) {
    float es    = scores[(size_t)e * Hh + h];
    float alpha = es / fmaxf(ss[(size_t)r * Hh + h], 1e-12f);
    float v     = V[(size_t)c * Dd + h * DHh + lane];
    atomicAdd(&agg[(size_t)r * Dd + h * DHh + lane], alpha * v);
  }
}

// ------------------------------- LayerNorm ---------------------------------
// out = LN(hin + add) * g + b ; one wave per row (D=256 -> 8 elems/lane)

__global__ __launch_bounds__(256) void ln_residual(
    const float* __restrict__ hin, const float* __restrict__ add,
    const float* __restrict__ g, const float* __restrict__ be,
    float* __restrict__ out, int n)
{
  int rowi = blockIdx.x * 8 + (threadIdx.x >> 5);
  if (rowi >= n) return;
  int lane = threadIdx.x & 31;
  size_t base = (size_t)rowi * Dd;
  float v[8];
  float sum = 0.0f;
  #pragma unroll
  for (int j = 0; j < 8; ++j) {
    int idx = lane + j * 32;
    v[j] = hin[base + idx] + add[base + idx];
    sum += v[j];
  }
  #pragma unroll
  for (int off = 16; off > 0; off >>= 1) sum += __shfl_xor(sum, off, 32);
  float mean = sum * (1.0f / Dd);
  float vs = 0.0f;
  #pragma unroll
  for (int j = 0; j < 8; ++j) { float d = v[j] - mean; vs += d * d; }
  #pragma unroll
  for (int off = 16; off > 0; off >>= 1) vs += __shfl_xor(vs, off, 32);
  float rstd = rsqrtf(vs * (1.0f / Dd) + 1e-5f);
  #pragma unroll
  for (int j = 0; j < 8; ++j) {
    int idx = lane + j * 32;
    out[base + idx] = (v[j] - mean) * rstd * g[idx] + be[idx];
  }
}

// ------------------------------- orchestration -----------------------------

extern "C" void kernel_launch(void* const* d_in, const int* in_sizes, int n_in,
                              void* d_out, int out_size, void* d_ws, size_t ws_size,
                              hipStream_t stream) {
  const float* x       = (const float*)d_in[0];
  const int*   row     = (const int*)  d_in[1];
  const int*   col     = (const int*)  d_in[2];
  const float* Wx      = (const float*)d_in[3];
  const float* bx      = (const float*)d_in[4];
  const float* deg_emb = (const float*)d_in[5];
  const float* sb      = (const float*)d_in[6];
  const float* Wq      = (const float*)d_in[7];
  const float* bq      = (const float*)d_in[8];
  const float* Wk      = (const float*)d_in[9];
  const float* bk      = (const float*)d_in[10];
  const float* Wv      = (const float*)d_in[11];
  const float* bv      = (const float*)d_in[12];
  const float* Wo      = (const float*)d_in[13];
  const float* bo      = (const float*)d_in[14];
  const float* ln1_g   = (const float*)d_in[15];
  const float* ln1_b   = (const float*)d_in[16];
  const float* ln2_g   = (const float*)d_in[17];
  const float* ln2_b   = (const float*)d_in[18];
  const float* W1      = (const float*)d_in[19];
  const float* b1      = (const float*)d_in[20];
  const float* W2      = (const float*)d_in[21];
  const float* b2      = (const float*)d_in[22];
  float* out = (float*)d_out;

  // workspace carve-out (256B aligned)
  char* ws = (char*)d_ws;
  size_t used = 0;
  auto carve = [&](size_t bytes) -> char* {
    char* p = ws + used;
    used += (bytes + 255) & ~(size_t)255;
    return p;
  };
  float*    deg    = (float*)   carve((size_t)Nn * 4);
  int*      bucket = (int*)     carve((size_t)Nn * 4);
  float*    h      = (float*)   carve((size_t)Nn * Dd * 4);
  float*    Qb     = (float*)   carve((size_t)Nn * Dd * 4);
  float*    Kb     = (float*)   carve((size_t)Nn * Dd * 4);   // contiguous after Qb
  float*    Vb     = (float*)   carve((size_t)Nn * Dd * 4);
  float*    agg    = (float*)   carve((size_t)Nn * Dd * 4);
  float*    tmp    = (float*)   carve((size_t)Nn * Dd * 4);
  float*    scores = (float*)   carve((size_t)ETOT * Hh * 4);
  float*    mx     = (float*)   carve((size_t)Nn * Hh * 4);
  float*    ss     = (float*)   carve((size_t)Nn * Hh * 4);
  _Float16* wf16   = (_Float16*)carve((size_t)(INC*Dd + Ll*(4*Dd*Dd + 2*Dd*DFFc)) * 2);
  float*    ffmid  = Qb;  // reuse Q+K region: Nn*DFF*4 == 2 * Nn*Dd*4
  if (used > ws_size) return;

  const dim3 blk(256);
  auto g1 = [](int n) { return dim3((unsigned)((n + 255) / 256)); };

  // degree + buckets
  fill_f32<<<g1(Nn), blk, 0, stream>>>(deg, 0.0f, Nn);
  deg_count<<<g1(Ee), blk, 0, stream>>>(row, deg, Ee);
  bucket_k<<<g1(Nn), blk, 0, stream>>>(deg, bucket, Nn);

  // weights -> f16 in workspace
  const size_t oWX = 0;
  const size_t perL = (size_t)4 * Dd * Dd + 2 * (size_t)Dd * DFFc;   // 524288
  auto wq16 = [&](int l){ return wf16 + INC*Dd + l*perL + 0*(size_t)Dd*Dd; };
  auto wk16 = [&](int l){ return wf16 + INC*Dd + l*perL + 1*(size_t)Dd*Dd; };
  auto wv16 = [&](int l){ return wf16 + INC*Dd + l*perL + 2*(size_t)Dd*Dd; };
  auto wo16 = [&](int l){ return wf16 + INC*Dd + l*perL + 3*(size_t)Dd*Dd; };
  auto w116 = [&](int l){ return wf16 + INC*Dd + l*perL + 4*(size_t)Dd*Dd; };
  auto w216 = [&](int l){ return wf16 + INC*Dd + l*perL + 4*(size_t)Dd*Dd + (size_t)Dd*DFFc; };
  f32_to_f16<<<g1(INC*Dd), blk, 0, stream>>>(Wx, wf16 + oWX, INC*Dd);
  for (int l = 0; l < Ll; ++l) {
    f32_to_f16<<<g1(Dd*Dd),   blk, 0, stream>>>(Wq + (size_t)l*Dd*Dd,   wq16(l), Dd*Dd);
    f32_to_f16<<<g1(Dd*Dd),   blk, 0, stream>>>(Wk + (size_t)l*Dd*Dd,   wk16(l), Dd*Dd);
    f32_to_f16<<<g1(Dd*Dd),   blk, 0, stream>>>(Wv + (size_t)l*Dd*Dd,   wv16(l), Dd*Dd);
    f32_to_f16<<<g1(Dd*Dd),   blk, 0, stream>>>(Wo + (size_t)l*Dd*Dd,   wo16(l), Dd*Dd);
    f32_to_f16<<<g1(Dd*DFFc), blk, 0, stream>>>(W1 + (size_t)l*Dd*DFFc, w116(l), Dd*DFFc);
    f32_to_f16<<<g1(Dd*DFFc), blk, 0, stream>>>(W2 + (size_t)l*Dd*DFFc, w216(l), Dd*DFFc);
  }

  const dim3 gD (Dd  / 64, (Nn + 127) / 128);   // (4, 157)
  const dim3 gFF(DFFc/ 64, (Nn + 127) / 128);   // (8, 157)
  const float scale = 0.17677669529663687f;     // 1/sqrt(32)

  // h0 = x @ Wx + bx + deg_emb[bucket]
  gemm_wmma<1><<<gD, blk, 0, stream>>>(x, wf16 + oWX, bx, h, Nn, INC, Dd, bucket, deg_emb);

  for (int l = 0; l < Ll; ++l) {
    gemm_wmma<0><<<gD, blk, 0, stream>>>(h, wq16(l), bq + l*Dd, Qb, Nn, Dd, Dd, nullptr, nullptr);
    gemm_wmma<0><<<gD, blk, 0, stream>>>(h, wk16(l), bk + l*Dd, Kb, Nn, Dd, Dd, nullptr, nullptr);
    gemm_wmma<0><<<gD, blk, 0, stream>>>(h, wv16(l), bv + l*Dd, Vb, Nn, Dd, Dd, nullptr, nullptr);

    fill_f32<<<g1(Nn*Hh), blk, 0, stream>>>(mx, -3.0e38f, Nn*Hh);
    fill_f32<<<g1(Nn*Hh), blk, 0, stream>>>(ss, 0.0f, Nn*Hh);
    fill_f32<<<g1(Nn*Dd), blk, 0, stream>>>(agg, 0.0f, Nn*Dd);

    edge_score<<<dim3((ETOT + 7) / 8), blk, 0, stream>>>(row, col, Qb, Kb, bucket,
                                                         sb + (size_t)l*Bb*Hh, scores, mx, scale);
    edge_exp<<<g1(ETOT*Hh), blk, 0, stream>>>(row, mx, scores, ss);
    edge_agg<<<dim3((ETOT + 7) / 8), blk, 0, stream>>>(row, col, Vb, scores, ss, agg);

    gemm_wmma<0><<<gD, blk, 0, stream>>>(agg, wo16(l), bo + l*Dd, tmp, Nn, Dd, Dd, nullptr, nullptr);
    ln_residual<<<dim3((Nn + 7) / 8), blk, 0, stream>>>(h, tmp, ln1_g + l*Dd, ln1_b + l*Dd, h, Nn);

    gemm_wmma<2><<<gFF, blk, 0, stream>>>(h, w116(l), b1 + l*DFFc, ffmid, Nn, Dd, DFFc, nullptr, nullptr);
    gemm_wmma<0><<<gD,  blk, 0, stream>>>(ffmid, w216(l), b2 + l*Dd, tmp, Nn, DFFc, Dd, nullptr, nullptr);

    float* lnout = (l == Ll - 1) ? out : h;
    ln_residual<<<dim3((Nn + 7) / 8), blk, 0, stream>>>(h, tmp, ln2_g + l*Dd, ln2_b + l*Dd, lnout, Nn);
  }
  (void)in_sizes; (void)n_in; (void)out_size;
}
